// KGAT_28432683499804
// MI455X (gfx1250) — matrix operand order repaired and checked by reading, not compiled, not run on hardware
//
#include <hip/hip_runtime.h>
#include <hip/hip_bf16.h>
#include <hip/hip_fp16.h>

typedef __attribute__((ext_vector_type(16))) _Float16 v16h;
typedef __attribute__((ext_vector_type(8)))  float    v8f;

// ---- order-preserving float<->uint encoding for atomicMax-based segment max ----
__device__ __forceinline__ unsigned fenc(float f) {
    unsigned u = __float_as_uint(f);
    return (u & 0x80000000u) ? ~u : (u | 0x80000000u);
}
__device__ __forceinline__ float fdec(unsigned u) {
    return (u & 0x80000000u) ? __uint_as_float(u & 0x7fffffffu) : __uint_as_float(~u);
}

// Branch-free tanh using the hardware transcendental pipe:
// tanh(x) = (e^{2x}-1)/(e^{2x}+1) = 1 - 2*rcp(e^{2x}+1)
// e^{2x}->inf => 1 ; e^{2x}->0 => -1. ~6 VALU/TRANS ops, no exec-mask branches.
__device__ __forceinline__ float tanh_fast(float x) {
    float t = __expf(2.0f * x);                       // v_mul + v_exp_f32
    return 1.0f - 2.0f * __builtin_amdgcn_rcpf(t + 1.0f);  // v_add + v_rcp_f32 + v_fma
}

// ---------------- setup kernels ----------------
__global__ void k_init(int* counts, int* cursors, unsigned* attmaxe, float* denom,
                       int nRel, int nNodes) {
    int i = blockIdx.x * blockDim.x + threadIdx.x;
    if (i < nRel)   { counts[i] = 0; cursors[i] = 0; }
    if (i < nNodes) { attmaxe[i] = 0u; denom[i] = 0.0f; }
}

__global__ void k_hist(const int* __restrict__ et, int* counts, int E) {
    int i = blockIdx.x * blockDim.x + threadIdx.x;
    if (i < E) atomicAdd(&counts[et[i]], 1);
}

__global__ void k_scan(const int* counts, int* offsets, int* cursors,
                       int* groupStart, int* total, int nRel) {
    if (blockIdx.x == 0 && threadIdx.x == 0) {
        int acc = 0, g = 0;
        for (int b = 0; b < nRel; ++b) {
            offsets[b] = acc; cursors[b] = acc; groupStart[b] = g;
            acc += counts[b];
            g   += (counts[b] + 15) >> 4;
        }
        groupStart[nRel] = g;
        total[0] = g;
    }
}

__global__ void k_scatter(const int* __restrict__ et, int* cursors, int* perm, int E) {
    int i = blockIdx.x * blockDim.x + threadIdx.x;
    if (i < E) { int p = atomicAdd(&cursors[et[i]], 1); perm[p] = i; }
}

__global__ void k_cvt_x(const float* __restrict__ x, _Float16* __restrict__ x16, long n) {
    long i = (long)blockIdx.x * blockDim.x + threadIdx.x;
    if (i < n) x16[i] = (_Float16)x[i];
}

// Pre-swizzle W_R[r] (64x64, KxN) into the exact 16x16x32 f16 B-matrix VGPR layout:
// block (r,kc,nt): 32 lanes x 16 contiguous f16; lane's value v maps to
// K = kc*32 + (v<8 ? (lane/16)*8 + v : 16 + (lane/16)*8 + v-8), N = nt*16 + (lane&15).
__global__ void k_pack_w(const float* __restrict__ W, _Float16* __restrict__ Bpk, int nRel) {
    int i = blockIdx.x * blockDim.x + threadIdx.x;
    if (i >= nRel * 4096) return;
    int t = i;
    int v    = t & 15; t >>= 4;
    int lane = t & 31; t >>= 5;
    int nt   = t & 3;  t >>= 2;
    int kc   = t & 1;  t >>= 1;
    int r    = t;
    int K = kc * 32 + (v < 8 ? (lane >> 4) * 8 + v : 16 + (lane >> 4) * 8 + (v - 8));
    int N = nt * 16 + (lane & 15);
    Bpk[i] = (_Float16)W[((long)r * 64 + K) * 64 + N];
}

// ---------------- main WMMA attention kernel ----------------
// One wave per group of 16 same-relation edges.
__global__ void __launch_bounds__(256)
k_attn(const _Float16* __restrict__ x16,
       const _Float16* __restrict__ Bpk,
       const float*    __restrict__ relE,
       const int* __restrict__ esrc, const int* __restrict__ edst,
       const int* __restrict__ perm,
       const int* __restrict__ offsets, const int* __restrict__ counts,
       const int* __restrict__ groupStart, const int* __restrict__ total,
       float* __restrict__ att, unsigned* __restrict__ attmaxe, int nRel)
{
    int gwave = (int)((blockIdx.x * blockDim.x + threadIdx.x) >> 5);
    int lane  = threadIdx.x & 31;
    if (gwave >= total[0]) return;

    int b = 0;
    while (b + 1 < nRel && gwave >= groupStart[b + 1]) ++b;
    int gi   = gwave - groupStart[b];
    int base = offsets[b] + gi * 16;
    int cnt  = counts[b] - gi * 16; if (cnt > 16) cnt = 16;

    int half = lane >> 4;
    int col  = lane & 15;

    // A-matrix row for this lane is edge m = lane%16 (safe-pad with edge 0 of group)
    int m   = col;
    int e_m = perm[base + (m < cnt ? m : 0)];
    long src = esrc[e_m];
    long dst = edst[e_m];
    const _Float16* rowS = x16 + src * 64;
    const _Float16* rowD = x16 + dst * 64;

    v8f Ct[4] = {}; // src path accumulators, 4 N-tiles of 16
    v8f Ch[4] = {}; // dst path accumulators

    #pragma unroll
    for (int kc = 0; kc < 2; ++kc) {
        // ISA 16-bit A 16x32 layout: v0..v3 = K = kc*32 + half*8 .. +7,
        //                            v4..v7 = K = kc*32 + 16 + half*8 .. +7
        union { v16h v; float4 f4[2]; } As, Ad;
        As.f4[0] = *(const float4*)(rowS + kc * 32 + half * 8);
        As.f4[1] = *(const float4*)(rowS + kc * 32 + 16 + half * 8);
        Ad.f4[0] = *(const float4*)(rowD + kc * 32 + half * 8);
        Ad.f4[1] = *(const float4*)(rowD + kc * 32 + 16 + half * 8);
        #pragma unroll
        for (int nt = 0; nt < 4; ++nt) {
            union { v16h v; float4 f4[2]; } Bv;
            const _Float16* bp = Bpk + ((((long)b * 2 + kc) * 4 + nt) * 32 + lane) * 16;
            Bv.f4[0] = *(const float4*)(bp);
            Bv.f4[1] = *(const float4*)(bp + 8);
            Ct[nt] = __builtin_amdgcn_wmma_f32_16x16x32_f16(
                false, As.v, false, Bv.v, (short)0, Ct[nt], false, false);
            Ch[nt] = __builtin_amdgcn_wmma_f32_16x16x32_f16(
                false, Ad.v, false, Bv.v, (short)0, Ch[nt], false, false);
        }
    }

    // relation embedding value for this lane's column in each N-tile
    float rel[4];
    #pragma unroll
    for (int nt = 0; nt < 4; ++nt) rel[nt] = relE[b * 64 + nt * 16 + col];

    // C layout: lane holds (M = half*8 + j, N = nt*16 + col) in Ct[nt][j].
    #pragma unroll
    for (int j = 0; j < 8; ++j) {
        float p = 0.f;
        #pragma unroll
        for (int nt = 0; nt < 4; ++nt)
            p += Ct[nt][j] * tanh_fast(Ch[nt][j] + rel[nt]);
        // reduce across the 16 lanes of this half (xor stays within half)
        #pragma unroll
        for (int off = 1; off < 16; off <<= 1) p += __shfl_xor(p, off, 32);
        int me = half * 8 + j;
        if (col == j && me < cnt) {
            int e2 = perm[base + me];
            att[e2] = p;
            atomicMax(attmaxe + edst[e2], fenc(p));
        }
    }
}

// ---------------- segmented softmax over edge_dst ----------------
__global__ void k_exp(const float* __restrict__ att, const unsigned* __restrict__ attmaxe,
                      const int* __restrict__ edst, float* __restrict__ exf,
                      float* __restrict__ denom, int E) {
    int i = blockIdx.x * blockDim.x + threadIdx.x;
    if (i < E) {
        int d = edst[i];
        float v = __expf(att[i] - fdec(attmaxe[d]));
        exf[i] = v;
        atomicAdd(&denom[d], v);
    }
}

__global__ void k_norm(const float* __restrict__ exf, const float* __restrict__ denom,
                       const int* __restrict__ edst, float* __restrict__ out, int E) {
    int i = blockIdx.x * blockDim.x + threadIdx.x;
    if (i < E) out[i] = exf[i] * __builtin_amdgcn_rcpf(denom[edst[i]]);
}

// ---------------- launch ----------------
extern "C" void kernel_launch(void* const* d_in, const int* in_sizes, int n_in,
                              void* d_out, int out_size, void* d_ws, size_t ws_size,
                              hipStream_t stream) {
    const float* x    = (const float*)d_in[0];
    const float* relE = (const float*)d_in[1];
    const float* W    = (const float*)d_in[2];
    const int*   esrc = (const int*)d_in[3];
    const int*   edst = (const int*)d_in[4];
    const int*   etyp = (const int*)d_in[5];

    int nNodes = in_sizes[0] / 64;
    int nRel   = in_sizes[1] / 64;
    int E      = in_sizes[3];

    // carve workspace (256-B aligned slices)
    char* p = (char*)d_ws;
    auto carve = [&](size_t bytes) -> char* {
        char* r = p; p += (bytes + 255) & ~(size_t)255; return r;
    };
    _Float16* x16     = (_Float16*)carve((size_t)nNodes * 64 * 2); // 15 MB, L2-resident
    _Float16* Bpk     = (_Float16*)carve((size_t)nRel * 4096 * 2); // 128 KB swizzled W
    int*      perm    = (int*)     carve((size_t)E * 4);
    float*    att     = (float*)   carve((size_t)E * 4);
    float*    exf     = (float*)   carve((size_t)E * 4);
    unsigned* attmaxe = (unsigned*)carve((size_t)nNodes * 4);
    float*    denom   = (float*)   carve((size_t)nNodes * 4);
    int*      meta    = (int*)     carve(128 * 4);
    int* counts = meta, *offsets = meta + 16, *cursors = meta + 32;
    int* groupStart = meta + 48, *total = meta + 80;

    const int T = 256;
    int nInit = nNodes > nRel ? nNodes : nRel;
    k_init   <<<(nInit + T - 1) / T, T, 0, stream>>>(counts, cursors, attmaxe, denom, nRel, nNodes);
    k_hist   <<<(E + T - 1) / T, T, 0, stream>>>(etyp, counts, E);
    k_scan   <<<1, 32, 0, stream>>>(counts, offsets, cursors, groupStart, total, nRel);
    k_scatter<<<(E + T - 1) / T, T, 0, stream>>>(etyp, cursors, perm, E);

    long nx = (long)nNodes * 64;
    k_cvt_x  <<<(int)((nx + T - 1) / T), T, 0, stream>>>(x, x16, nx);
    k_pack_w <<<(nRel * 4096 + T - 1) / T, T, 0, stream>>>(W, Bpk, nRel);

    int maxGroups = (E + 15) / 16 + nRel;       // upper bound on total wave-groups
    int blocks = (maxGroups * 32 + T - 1) / T;  // 8 waves / block
    k_attn<<<blocks, T, 0, stream>>>(x16, Bpk, relE, esrc, edst, perm,
                                     offsets, counts, groupStart, total,
                                     att, attmaxe, nRel);

    k_exp <<<(E + T - 1) / T, T, 0, stream>>>(att, attmaxe, edst, exf, denom, E);
    k_norm<<<(E + T - 1) / T, T, 0, stream>>>(exf, denom, edst, (float*)d_out, E);
}
